// DCRNN_40819369181336
// MI455X (gfx1250) — matrix operand order repaired and checked by reading, not compiled, not run on hardware
//
#include <hip/hip_runtime.h>
#include <hip/hip_bf16.h>

// ---- problem constants (from reference) ----
#define BATCH   32
#define TSTEPS  12
#define NSEQ    2048
#define DIN     2
#define HDIM    64
#define NROWS   (BATCH * NSEQ)      // 65536 independent rows
#define ROWS_PER_BLOCK 64
#define WAVES_PER_BLOCK 4           // wave32: 4 waves * 16 rows = 64 rows

typedef _Float16 v16h __attribute__((ext_vector_type(16)));
typedef float    v8f  __attribute__((ext_vector_type(8)));

union V16HU { v16h v; unsigned u[8]; };

// A operand 16x32 f16 (doc layout): lane holds row M=lane%16; K striping:
// K = (v/4)*16 + (lane/16)*8 + (v%4)*2 + j
__device__ __forceinline__ v16h lds_loadA(const _Float16* __restrict__ s,
                                          int kc, int m, int hi) {
  V16HU r;
  const unsigned* sp = (const unsigned*)s;
  #pragma unroll
  for (int v = 0; v < 8; ++v) {
    int kl = ((v >> 2) << 4) + (hi << 3) + ((v & 3) << 1);
    int k  = (kc << 5) + kl;
    r.u[v] = sp[(m * 66 + k) >> 1];           // stride 66 f16 (33 dwords, conflict-free)
  }
  return r.v;
}

// B operand 32x16 f16 (sparse-table analog): lane holds col N=lane%16;
// K = (v/4)*16 + (lane/16)*8 + (v%4)*2 + j ; weights stored transposed [N][K] in LDS.
__device__ __forceinline__ v16h lds_loadB(const _Float16* __restrict__ wt,
                                          int stride, int nt, int kc,
                                          int n16, int hi) {
  V16HU r;
  const unsigned* sp = (const unsigned*)wt;
  #pragma unroll
  for (int v = 0; v < 8; ++v) {
    int kl = ((v >> 2) << 4) + (hi << 3) + ((v & 3) << 1);
    int k  = (kc << 5) + kl;
    r.u[v] = sp[(((nt << 4) + n16) * stride + k) >> 1];
  }
  return r.v;
}

// C/D layout: VGPR i, lane L -> (M = i + 8*(L/16), N = L%16); spill f32->f16 to LDS tile.
__device__ __forceinline__ void lds_storeC(_Float16* __restrict__ s,
                                           const v8f* hv, int n16, int hi) {
  #pragma unroll
  for (int nt = 0; nt < 4; ++nt)
    #pragma unroll
    for (int i = 0; i < 8; ++i) {
      int m = i + (hi << 3);
      s[m * 66 + (nt << 4) + n16] = (_Float16)hv[nt][i];
    }
}

__device__ __forceinline__ float fast_sigmoid(float x) {
  return 1.0f / (1.0f + __expf(-x));
}
__device__ __forceinline__ float fast_tanh(float x) {
  x = fminf(fmaxf(x, -15.0f), 15.0f);
  float e = __expf(2.0f * x);
  return (e - 1.0f) / (e + 1.0f);
}

__global__ __launch_bounds__(128, 1) void dcrnn_fused_kernel(
    const float* __restrict__ x,
    const float* __restrict__ Wu0, const float* __restrict__ bu0,
    const float* __restrict__ Wc0, const float* __restrict__ bc0,
    const float* __restrict__ Wu1, const float* __restrict__ bu1,
    const float* __restrict__ Wc1, const float* __restrict__ bc1,
    const float* __restrict__ Wo,  const float* __restrict__ bo,
    float* __restrict__ out)
{
  // ---- LDS (~60 KB total) ----
  __shared__ _Float16 sh_scr[WAVES_PER_BLOCK][16 * 66];  // per-wave C<->A transpose scratch
  __shared__ _Float16 sh_wu0T[64 * 66];                  // Wu0[2:66,:]^T  [N][K=64]
  __shared__ _Float16 sh_wc0T[64 * 66];
  __shared__ _Float16 sh_wu1T[64 * 130];                 // Wu1^T          [N][K=128]
  __shared__ _Float16 sh_wc1T[64 * 130];
  __shared__ float sh_wxu[2 * 64];                       // x-part rows of Wu0 / Wc0
  __shared__ float sh_wxc[2 * 64];
  __shared__ float sh_bu0[64], sh_bc0[64], sh_bu1[64], sh_bc1[64];
  __shared__ float sh_wo[64];
  __shared__ float sh_bo;
  __shared__ float sh_x[ROWS_PER_BLOCK * DIN];           // staged x_t tile

  const int tid   = threadIdx.x;
  const int w     = tid >> 5;
  const int lane  = tid & 31;
  const int n16   = lane & 15;   // also the A-operand row within the wave tile
  const int hi    = lane >> 4;
  const int wrow0 = w << 4;

  // ---- one-time weight transpose + f32->f16 convert into LDS ----
  for (int idx = tid; idx < 64 * 64; idx += 128) {
    int n = idx >> 6, k = idx & 63;
    sh_wu0T[n * 66 + k] = (_Float16)Wu0[(k + 2) * 64 + n];   // skip x rows 0..1
    sh_wc0T[n * 66 + k] = (_Float16)Wc0[(k + 2) * 64 + n];
  }
  for (int idx = tid; idx < 64 * 128; idx += 128) {
    int n = idx >> 7, k = idx & 127;
    sh_wu1T[n * 130 + k] = (_Float16)Wu1[k * 64 + n];
    sh_wc1T[n * 130 + k] = (_Float16)Wc1[k * 64 + n];
  }
  { sh_wxu[tid] = Wu0[tid]; sh_wxc[tid] = Wc0[tid]; }        // rows 0..1 (128 elems)
  if (tid < 64) {
    sh_bu0[tid] = bu0[tid]; sh_bc0[tid] = bc0[tid];
    sh_bu1[tid] = bu1[tid]; sh_bc1[tid] = bc1[tid];
    sh_wo[tid]  = Wo[tid];
  }
  if (tid == 0) sh_bo = bo[0];
  __syncthreads();

  // ---- per-lane constant preloads (N-column dependent) ----
  float bu0_r[4], bc0_r[4], bu1_r[4], bc1_r[4], wxu_r[4][2], wxc_r[4][2];
  #pragma unroll
  for (int nt = 0; nt < 4; ++nt) {
    int col = (nt << 4) + n16;
    bu0_r[nt] = sh_bu0[col]; bc0_r[nt] = sh_bc0[col];
    bu1_r[nt] = sh_bu1[col]; bc1_r[nt] = sh_bc1[col];
    wxu_r[nt][0] = sh_wxu[col]; wxu_r[nt][1] = sh_wxu[64 + col];
    wxc_r[nt][0] = sh_wxc[col]; wxc_r[nt][1] = sh_wxc[64 + col];
  }

  // ---- hidden state lives in VGPRs (C-layout) across all timesteps ----
  v8f h0v[4], h1v[4];
  #pragma unroll
  for (int nt = 0; nt < 4; ++nt)
    #pragma unroll
    for (int i = 0; i < 8; ++i) { h0v[nt][i] = 0.0f; h1v[nt][i] = 0.0f; }

  const int row0 = blockIdx.x * ROWS_PER_BLOCK;
  const int bb   = row0 / NSEQ;
  const int nn0  = row0 % NSEQ;
  _Float16* scr  = &sh_scr[w][0];

  #pragma unroll 1
  for (int t = 0; t < TSTEPS; ++t) {
    // ---- stage x_t tile (64 rows x 2) ----
    __syncthreads();
    int xbase = ((bb * TSTEPS + t) * NSEQ + nn0) * DIN;
    sh_x[tid] = x[xbase + tid];
    if (t + 1 < TSTEPS)
      __builtin_prefetch(&x[xbase + NSEQ * DIN + tid], 0, 0);  // global_prefetch_b8
    __syncthreads();

    // ================= layer 0:  K = [x(2) | h0(64)] =================
    lds_storeC(scr, h0v, n16, hi);                 // h0 (prev) -> A layout
    v16h a0 = lds_loadA(scr, 0, n16, hi);
    v16h a1 = lds_loadA(scr, 1, n16, hi);

    float xv0[8], xv1[8];
    #pragma unroll
    for (int i = 0; i < 8; ++i) {
      int rl = wrow0 + i + (hi << 3);
      xv0[i] = sh_x[rl * 2 + 0];
      xv1[i] = sh_x[rl * 2 + 1];
    }

    #pragma unroll
    for (int nt = 0; nt < 4; ++nt) {
      v8f ua, ca;
      #pragma unroll
      for (int i = 0; i < 8; ++i) {                // bias + tiny K=2 x-part via FMA
        ua[i] = bu0_r[nt] + xv0[i] * wxu_r[nt][0] + xv1[i] * wxu_r[nt][1];
        ca[i] = bc0_r[nt] + xv0[i] * wxc_r[nt][0] + xv1[i] * wxc_r[nt][1];
      }
      ua = __builtin_amdgcn_wmma_f32_16x16x32_f16(false, a0, false,
             lds_loadB(sh_wu0T, 66, nt, 0, n16, hi), (short)0, ua, false, false);
      ua = __builtin_amdgcn_wmma_f32_16x16x32_f16(false, a1, false,
             lds_loadB(sh_wu0T, 66, nt, 1, n16, hi), (short)0, ua, false, false);
      ca = __builtin_amdgcn_wmma_f32_16x16x32_f16(false, a0, false,
             lds_loadB(sh_wc0T, 66, nt, 0, n16, hi), (short)0, ca, false, false);
      ca = __builtin_amdgcn_wmma_f32_16x16x32_f16(false, a1, false,
             lds_loadB(sh_wc0T, 66, nt, 1, n16, hi), (short)0, ca, false, false);
      #pragma unroll
      for (int i = 0; i < 8; ++i) {                // GRU blend (r gate is dead code)
        float u = fast_sigmoid(ua[i]);
        float c = fast_tanh(ca[i]);
        h0v[nt][i] = u * h0v[nt][i] + (1.0f - u) * c;
      }
    }

    // ================= layer 1:  K = [h0(64) | h1(64)] =================
    v16h a[4];
    lds_storeC(scr, h0v, n16, hi);                 // new h0 -> A chunks 0,1
    a[0] = lds_loadA(scr, 0, n16, hi);
    a[1] = lds_loadA(scr, 1, n16, hi);
    lds_storeC(scr, h1v, n16, hi);                 // prev h1 -> A chunks 2,3
    a[2] = lds_loadA(scr, 0, n16, hi);
    a[3] = lds_loadA(scr, 1, n16, hi);

    #pragma unroll
    for (int nt = 0; nt < 4; ++nt) {
      v8f ua, ca;
      #pragma unroll
      for (int i = 0; i < 8; ++i) { ua[i] = bu1_r[nt]; ca[i] = bc1_r[nt]; }
      #pragma unroll
      for (int kc = 0; kc < 4; ++kc)
        ua = __builtin_amdgcn_wmma_f32_16x16x32_f16(false, a[kc], false,
               lds_loadB(sh_wu1T, 130, nt, kc, n16, hi), (short)0, ua, false, false);
      #pragma unroll
      for (int kc = 0; kc < 4; ++kc)
        ca = __builtin_amdgcn_wmma_f32_16x16x32_f16(false, a[kc], false,
               lds_loadB(sh_wc1T, 130, nt, kc, n16, hi), (short)0, ca, false, false);
      #pragma unroll
      for (int i = 0; i < 8; ++i) {
        float u = fast_sigmoid(ua[i]);
        float c = fast_tanh(ca[i]);
        h1v[nt][i] = u * h1v[nt][i] + (1.0f - u) * c;
      }
    }
  }

  // ---- output: out = h1 @ Wo + bo  (O=1 -> per-row dot of 64) ----
  lds_storeC(scr, h1v, n16, hi);
  if (lane < 16) {
    float acc = sh_bo;
    #pragma unroll 8
    for (int k = 0; k < 64; ++k)
      acc += (float)scr[lane * 66 + k] * sh_wo[k];
    out[row0 + wrow0 + lane] = acc;
  }
}

extern "C" void kernel_launch(void* const* d_in, const int* in_sizes, int n_in,
                              void* d_out, int out_size, void* d_ws, size_t ws_size,
                              hipStream_t stream) {
  (void)in_sizes; (void)n_in; (void)out_size; (void)d_ws; (void)ws_size;
  // setup_inputs order:
  // 0:x 1:supports 2:Wr0 3:br0 4:Wu0 5:bu0 6:Wc0 7:bc0 8:Wd0 9:bd0
  // 10:Wr1 11:br1 12:Wu1 13:bu1 14:Wc1 15:bc1 16:Wd1 17:bd1 18:Wo 19:bo
  const float* x   = (const float*)d_in[0];
  const float* Wu0 = (const float*)d_in[4];
  const float* bu0 = (const float*)d_in[5];
  const float* Wc0 = (const float*)d_in[6];
  const float* bc0 = (const float*)d_in[7];
  const float* Wu1 = (const float*)d_in[12];
  const float* bu1 = (const float*)d_in[13];
  const float* Wc1 = (const float*)d_in[14];
  const float* bc1 = (const float*)d_in[15];
  const float* Wo  = (const float*)d_in[18];
  const float* bo  = (const float*)d_in[19];
  float* out = (float*)d_out;

  dim3 grid(NROWS / ROWS_PER_BLOCK);   // 1024 blocks
  dim3 block(128);                     // 4 wave32 waves, 16 rows each
  dcrnn_fused_kernel<<<grid, block, 0, stream>>>(
      x, Wu0, bu0, Wc0, bc0, Wu1, bu1, Wc1, bc1, Wo, bo, out);
}
